// PoyoHomogenizer_35983236006162
// MI455X (gfx1250) — matrix operand order repaired and checked by reading, not compiled, not run on hardware
//
#include <hip/hip_runtime.h>
#include <hip/hip_bf16.h>
#include <cstdint>
#include <cstddef>

#define B_    16
#define N_IN  4096
#define N_LAT 512
#define DIM   512
#define HEADS 8
#define DHEAD 64
#define INNER 512
#define FF    2048

#define BM 128
#define BN 64
#define BK 32
#define LDK 40   // padded LDS row stride in halves (80B, keeps 16B alignment)

typedef __bf16 bf16_t;
typedef __bf16 v16bf __attribute__((ext_vector_type(16)));
typedef __bf16 v8bf  __attribute__((ext_vector_type(8)));
typedef float  v8f   __attribute__((ext_vector_type(8)));
typedef int    i4v   __attribute__((ext_vector_type(4)));
typedef uint32_t su4 __attribute__((ext_vector_type(4)));
typedef uint32_t su8 __attribute__((ext_vector_type(8)));

// ---- CDNA5 async / tensor data-mover plumbing -----------------------------
#if __has_builtin(__builtin_amdgcn_global_load_async_to_lds_b128)
#define HAVE_ASYNC_LDS 1
#endif

__device__ __forceinline__ void cp_async16(bf16_t* lds, const bf16_t* g) {
#ifdef HAVE_ASYNC_LDS
  __builtin_amdgcn_global_load_async_to_lds_b128(
      (__attribute__((address_space(1))) i4v*)(uintptr_t)g,
      (__attribute__((address_space(3))) i4v*)(uintptr_t)lds, 0, 0);
#else
  *(v8bf*)lds = *(const v8bf*)g;
#endif
}

#if __has_builtin(__builtin_amdgcn_s_wait_asynccnt)
#define WAIT_ASYNC(n) __builtin_amdgcn_s_wait_asynccnt(n)
#else
#define WAIT_ASYNC(n) asm volatile("s_wait_asynccnt %0" ::"i"(n) : "memory")
#endif

#if __has_builtin(__builtin_amdgcn_s_wait_tensorcnt)
#define WAIT_TENSOR(n) __builtin_amdgcn_s_wait_tensorcnt(n)
#else
#define WAIT_TENSOR(n) asm volatile("s_wait_tensorcnt %0" ::"i"(n) : "memory")
#endif

// TDM: DMA a 2D bf16 tile (rows x 32 halves, row stride K elements) from
// global memory into LDS, with LDS padding so rows land at an 80B stride
// (pad_interval = 16 DWORDs -> code 3, pad_amount = 4 DWORDs -> code 3).
__device__ __forceinline__ void tdm_load_tile(uint32_t lds_byte_addr,
                                              const bf16_t* gsrc,
                                              uint32_t rows, uint32_t K) {
  uint64_t ga = (uint64_t)(uintptr_t)gsrc;
  su4 g0;
  g0.x = 1u;                                   // count = 1 valid descriptor
  g0.y = lds_byte_addr;                        // lds_addr
  g0.z = (uint32_t)ga;                         // global_addr[31:0]
  g0.w = ((uint32_t)(ga >> 32) & 0x01FFFFFFu) | (2u << 30);  // addr[56:32]|type=2
  su8 g1;
  g1.s0 = (1u << 16) | (1u << 20) | (3u << 22) | (3u << 25); // 2B elems, pad en
  g1.s1 = (32u & 0xFFFFu) << 16;               // tensor_dim0 = 32 (low16)
  g1.s2 = (rows & 0xFFFFu) << 16;              // tensor_dim0 hi | tensor_dim1
  g1.s3 = (32u << 16);                         // tensor_dim1 hi | tile_dim0=32
  g1.s4 = rows;                                // tile_dim1 = rows, tile_dim2=0
  g1.s5 = K;                                   // tensor_dim0_stride low32
  g1.s6 = 0u;
  g1.s7 = 0u;
  su4 g2 = {0u, 0u, 0u, 0u};
  su4 g3 = {0u, 0u, 0u, 0u};
  asm volatile("tensor_load_to_lds %0, %1, %2, %3"
               ::"s"(g0), "s"(g1), "s"(g2), "s"(g3)
               : "memory");
}

__device__ __forceinline__ v8f zero_v8f() {
  v8f z = {0.f, 0.f, 0.f, 0.f, 0.f, 0.f, 0.f, 0.f};
  return z;
}

// Assemble a 16x32 A-fragment / 32x16 B-fragment (16 halves per lane) from two
// contiguous 16-byte chunks (per the CDNA5 16-bit WMMA VGPR layouts).
__device__ __forceinline__ v16bf load_frag(const bf16_t* lo, const bf16_t* hi) {
  v8bf l = *(const v8bf*)lo;
  v8bf h = *(const v8bf*)hi;
  return __builtin_shufflevector(l, h, 0, 1, 2, 3, 4, 5, 6, 7,
                                 8, 9, 10, 11, 12, 13, 14, 15);
}

__device__ __forceinline__ float inv_freq_of(int j) {
  if (j >= 16) return 0.0f;
  return 6.28318530718f / (1e-4f * __powf(40000.0f, (float)j * (1.0f / 16.0f)));
}

// ---------------------------------------------------------------------------
// Weight transpose + fp32 -> bf16  (dst is N x K row-major = B^T layout)
// ---------------------------------------------------------------------------
__global__ void transpose_to_bf16(const float* __restrict__ src,
                                  bf16_t* __restrict__ dst, int K, int N) {
  size_t i = (size_t)blockIdx.x * blockDim.x + threadIdx.x;
  size_t total = (size_t)K * N;
  if (i >= total) return;
  int k = (int)(i / N), n = (int)(i % N);
  dst[(size_t)n * K + k] = (bf16_t)src[i];
}

// ---------------------------------------------------------------------------
// LayerNorm over rows of 512, fp32 in -> bf16 out. One block (256 thr) per row.
// ---------------------------------------------------------------------------
__global__ __launch_bounds__(256)
void layernorm_to_bf16(const float* __restrict__ x, const float* __restrict__ w,
                       const float* __restrict__ b, bf16_t* __restrict__ y) {
  __shared__ float red[16];
  int row = blockIdx.x;
  int tid = threadIdx.x;
  const float* xr = x + (size_t)row * DIM;
  float x0 = xr[tid], x1 = xr[tid + 256];
  float s = x0 + x1, s2 = x0 * x0 + x1 * x1;
  for (int off = 16; off > 0; off >>= 1) {
    s  += __shfl_xor(s, off, 32);
    s2 += __shfl_xor(s2, off, 32);
  }
  int wv = tid >> 5, ln = tid & 31;
  if (ln == 0) { red[wv] = s; red[8 + wv] = s2; }
  __syncthreads();
  float ts = 0.f, ts2 = 0.f;
  for (int i = 0; i < 8; i++) { ts += red[i]; ts2 += red[8 + i]; }
  float mu  = ts * (1.0f / DIM);
  float var = ts2 * (1.0f / DIM) - mu * mu;
  float rstd = rsqrtf(var + 1e-5f);
  bf16_t* yr = y + (size_t)row * DIM;
  yr[tid]       = (bf16_t)((x0 - mu) * rstd * w[tid] + b[tid]);
  yr[tid + 256] = (bf16_t)((x1 - mu) * rstd * w[tid + 256] + b[tid + 256]);
}

// ---------------------------------------------------------------------------
// Generic bf16 WMMA GEMM, double-buffered LDS:
//   A tile (128x32) via per-lane async global->LDS copies (ASYNCcnt),
//   B tile (64x32)  via Tensor Data Mover, issued by wave 0 (TENSORcnt).
// Epilogue modes:
//   0: fp32 out = acc (+bias[n]) (+res[m*N+n])           -> Cout
//   1: rotary(scale*acc) with ts[m], split heads         -> OutX (B,H,NLAT,DH)
//   2: rotary(acc) with ts[m]; n<INNER -> K (B,H,NIN,DH) -> OutX
//                              n>=INNER -> V^T (B,H,DH,NIN) -> OutV
// ---------------------------------------------------------------------------
__global__ __launch_bounds__(256)
void gemm_bf16(const bf16_t* __restrict__ A, const bf16_t* __restrict__ BT,
               int M, int N, int K, int mode,
               const float* __restrict__ bias, const float* res,
               float* Cout, const float* __restrict__ ts,
               bf16_t* __restrict__ OutX, bf16_t* __restrict__ OutV) {
  __shared__ __align__(16) bf16_t As[2][BM][LDK];
  __shared__ __align__(16) bf16_t Bs[2][BN][LDK];
  int tid = threadIdx.x;
  int wv = tid >> 5, lane = tid & 31;
  int grp = lane >> 4, lr = lane & 15;
  int wm = wv >> 1, wn = wv & 1;
  int m0 = blockIdx.y * BM, n0 = blockIdx.x * BN;

  int ar = tid >> 2, ac = (tid & 3) << 3;  // each thread: 2 A chunks of 8 halves

  v8f acc[2][2];
  for (int mi = 0; mi < 2; mi++)
    for (int ni = 0; ni < 2; ni++) acc[mi][ni] = zero_v8f();

  const int T = K / BK;
  auto issueA = [&](int t, int buf) {
    int k0 = t * BK;
    cp_async16(&As[buf][ar][ac], A + (size_t)(m0 + ar) * K + k0 + ac);
    cp_async16(&As[buf][64 + ar][ac], A + (size_t)(m0 + 64 + ar) * K + k0 + ac);
  };
  auto issueB = [&](int t, int buf) {
    if (wv == 0) {
      uint32_t ldsb = (uint32_t)(uintptr_t)&Bs[buf][0][0];
      tdm_load_tile(ldsb, BT + (size_t)n0 * K + t * BK, BN, (uint32_t)K);
    }
  };

  issueA(0, 0);
  issueB(0, 0);
  for (int t = 0; t < T; t++) {
    int buf = t & 1;
    if (t + 1 < T) {
      issueA(t + 1, buf ^ 1);
      issueB(t + 1, buf ^ 1);
      WAIT_ASYNC(2);                 // tile t's 2 async copies done (in-order)
      if (wv == 0) WAIT_TENSOR(1);   // tile t's TDM done
    } else {
      WAIT_ASYNC(0);
      if (wv == 0) WAIT_TENSOR(0);
    }
    __syncthreads();

    v16bf afr[2], bfr[2];
    for (int mi = 0; mi < 2; mi++) {
      int r = wm * 32 + mi * 16 + lr;
      afr[mi] = load_frag(&As[buf][r][grp * 8], &As[buf][r][16 + grp * 8]);
    }
    for (int ni = 0; ni < 2; ni++) {
      int r = wn * 32 + ni * 16 + lr;
      bfr[ni] = load_frag(&Bs[buf][r][grp * 8], &Bs[buf][r][16 + grp * 8]);
    }
    for (int mi = 0; mi < 2; mi++)
      for (int ni = 0; ni < 2; ni++)
        acc[mi][ni] = __builtin_amdgcn_wmma_f32_16x16x32_bf16(
            false, afr[mi], false, bfr[ni], (short)0, acc[mi][ni], false, false);
    __syncthreads();
  }

  // Epilogue (D layout: lane holds column n = tile + lr, rows r + 8*grp).
  for (int mi = 0; mi < 2; mi++) {
    for (int ni = 0; ni < 2; ni++) {
      int col = n0 + wn * 32 + ni * 16 + lr;
      for (int r = 0; r < 8; r++) {
        int m = m0 + wm * 32 + mi * 16 + r + 8 * grp;
        float val = acc[mi][ni][r];
        if (mode == 0) {
          float o = val;
          if (bias) o += bias[col];
          if (res) o += res[(size_t)m * N + col];
          Cout[(size_t)m * N + col] = o;
        } else if (mode == 1) {
          float v = val * 0.125f;  // DHEAD^-0.5 folded into q
          float pv = __shfl_xor(v, 1, 32);
          int h = col >> 6, d = col & 63, j = d >> 1;
          float c, s;
          __sincosf(ts[m] * inv_freq_of(j), &s, &c);
          float o = (d & 1) ? (v * c + pv * s) : (v * c - pv * s);
          int bb = m >> 9, qi = m & (N_LAT - 1);
          OutX[(((size_t)bb * HEADS + h) * N_LAT + qi) * DHEAD + d] = (bf16_t)o;
        } else {  // mode 2: k / v with rotary; v stored transposed
          float v = val;
          float pv = __shfl_xor(v, 1, 32);
          int bb = m >> 12, tok = m & (N_IN - 1);
          int isV = col >= INNER;
          int cc = isV ? col - INNER : col;
          int h = cc >> 6, d = cc & 63, j = d >> 1;
          float c, s;
          __sincosf(ts[m] * inv_freq_of(j), &s, &c);
          float o = (d & 1) ? (v * c + pv * s) : (v * c - pv * s);
          if (isV)
            OutV[(((size_t)bb * HEADS + h) * DHEAD + d) * N_IN + tok] = (bf16_t)o;
          else
            OutX[(((size_t)bb * HEADS + h) * N_IN + tok) * DHEAD + d] = (bf16_t)o;
        }
      }
    }
  }
}

// ---------------------------------------------------------------------------
// FFN1 (dual B tiles for the gated halves), double-buffered async staging.
// gated = (lnf@w1[:, :FF] + b1[:FF]) * gelu(lnf@w1[:, FF:] + b1[FF:])
// ---------------------------------------------------------------------------
__global__ __launch_bounds__(256)
void gemm_ffn1_gated(const bf16_t* __restrict__ A, const bf16_t* __restrict__ W1T,
                     const float* __restrict__ b1, bf16_t* __restrict__ gated,
                     int M, int K) {
  __shared__ __align__(16) bf16_t As[2][BM][LDK];
  __shared__ __align__(16) bf16_t Bx[2][BN][LDK];
  __shared__ __align__(16) bf16_t Bg[2][BN][LDK];
  int tid = threadIdx.x;
  int wv = tid >> 5, lane = tid & 31;
  int grp = lane >> 4, lr = lane & 15;
  int wm = wv >> 1, wn = wv & 1;
  int m0 = blockIdx.y * BM, n0 = blockIdx.x * BN;
  int ar = tid >> 2, ac = (tid & 3) << 3;

  v8f ax[2][2], ag[2][2];
  for (int mi = 0; mi < 2; mi++)
    for (int ni = 0; ni < 2; ni++) { ax[mi][ni] = zero_v8f(); ag[mi][ni] = zero_v8f(); }

  const int T = K / BK;
  auto issue = [&](int t, int buf) {
    int k0 = t * BK;
    cp_async16(&As[buf][ar][ac], A + (size_t)(m0 + ar) * K + k0 + ac);
    cp_async16(&As[buf][64 + ar][ac], A + (size_t)(m0 + 64 + ar) * K + k0 + ac);
    cp_async16(&Bx[buf][ar & 63][ac],
               W1T + (size_t)(n0 + (ar & 63)) * K + k0 + ac);
    cp_async16(&Bg[buf][ar & 63][ac],
               W1T + (size_t)(n0 + FF + (ar & 63)) * K + k0 + ac);
  };

  issue(0, 0);
  for (int t = 0; t < T; t++) {
    int buf = t & 1;
    if (t + 1 < T) {
      issue(t + 1, buf ^ 1);
      WAIT_ASYNC(4);
    } else {
      WAIT_ASYNC(0);
    }
    __syncthreads();
    v16bf afr[2], bxf[2], bgf[2];
    for (int mi = 0; mi < 2; mi++) {
      int r = wm * 32 + mi * 16 + lr;
      afr[mi] = load_frag(&As[buf][r][grp * 8], &As[buf][r][16 + grp * 8]);
    }
    for (int ni = 0; ni < 2; ni++) {
      int r = wn * 32 + ni * 16 + lr;
      bxf[ni] = load_frag(&Bx[buf][r][grp * 8], &Bx[buf][r][16 + grp * 8]);
      bgf[ni] = load_frag(&Bg[buf][r][grp * 8], &Bg[buf][r][16 + grp * 8]);
    }
    for (int mi = 0; mi < 2; mi++)
      for (int ni = 0; ni < 2; ni++) {
        ax[mi][ni] = __builtin_amdgcn_wmma_f32_16x16x32_bf16(
            false, afr[mi], false, bxf[ni], (short)0, ax[mi][ni], false, false);
        ag[mi][ni] = __builtin_amdgcn_wmma_f32_16x16x32_bf16(
            false, afr[mi], false, bgf[ni], (short)0, ag[mi][ni], false, false);
      }
    __syncthreads();
  }

  for (int mi = 0; mi < 2; mi++)
    for (int ni = 0; ni < 2; ni++) {
      int col = n0 + wn * 32 + ni * 16 + lr;
      for (int r = 0; r < 8; r++) {
        int m = m0 + wm * 32 + mi * 16 + r + 8 * grp;
        float hx = ax[mi][ni][r] + b1[col];
        float hg = ag[mi][ni][r] + b1[col + FF];
        float gl = 0.5f * hg * (1.0f + erff(hg * 0.70710678f));  // exact gelu
        gated[(size_t)m * FF + col] = (bf16_t)(hx * gl);
      }
    }
}

// ---------------------------------------------------------------------------
// Flash cross-attention. Grid (NLAT/128, B*H); 8 waves, 16 queries per wave.
// Streams 32-key tiles: 4 QK^T WMMAs + 4 PV WMMAs per tile; online softmax
// with 16-lane shfl reductions; inverse rotary fused into output store.
// ---------------------------------------------------------------------------
__global__ __launch_bounds__(256)
void attention_kernel(const bf16_t* __restrict__ Q, const bf16_t* __restrict__ Kb,
                      const bf16_t* __restrict__ Vt,
                      const unsigned char* __restrict__ mask,
                      const float* __restrict__ lat_ts, bf16_t* __restrict__ Out) {
  __shared__ __align__(16) bf16_t Ps[8][16][40];
  int tid = threadIdx.x, wv = tid >> 5, lane = tid & 31;
  int grp = lane >> 4, lr = lane & 15;
  int bh = blockIdx.y, bb = bh >> 3, h = bh & 7;
  int q0 = blockIdx.x * 128 + wv * 16;

  const bf16_t* qrow = Q + ((size_t)bh * N_LAT + q0 + lr) * DHEAD;
  v16bf qa0 = load_frag(qrow + grp * 8, qrow + 16 + grp * 8);
  v16bf qa1 = load_frag(qrow + 32 + grp * 8, qrow + 48 + grp * 8);

  float rmax[8], rsum[8];
  for (int r = 0; r < 8; r++) { rmax[r] = -1e38f; rsum[r] = 0.0f; }
  v8f o[4];
  for (int t = 0; t < 4; t++) o[t] = zero_v8f();

  const unsigned char* mrow = mask + (size_t)bb * N_IN;
  for (int kt = 0; kt < N_IN; kt += 32) {
    if (kt + 32 < N_IN) {  // prefetch next K/V tiles into cache
      __builtin_prefetch(Kb + ((size_t)bh * N_IN + kt + 32 + lane) * DHEAD, 0, 1);
      __builtin_prefetch(Vt + ((size_t)bh * DHEAD + lane * 2) * N_IN + kt + 32, 0, 1);
    }
    v8f sc[2];
    for (int j = 0; j < 2; j++) {
      const bf16_t* krow = Kb + ((size_t)bh * N_IN + kt + j * 16 + lr) * DHEAD;
      v16bf kb0 = load_frag(krow + grp * 8, krow + 16 + grp * 8);
      v16bf kb1 = load_frag(krow + 32 + grp * 8, krow + 48 + grp * 8);
      v8f z = zero_v8f();
      z = __builtin_amdgcn_wmma_f32_16x16x32_bf16(false, qa0, false, kb0,
                                                  (short)0, z, false, false);
      z = __builtin_amdgcn_wmma_f32_16x16x32_bf16(false, qa1, false, kb1,
                                                  (short)0, z, false, false);
      bool mk = mrow[kt + j * 16 + lr] != 0;
      for (int r = 0; r < 8; r++) sc[j][r] = mk ? z[r] : -1e30f;
    }
    for (int r = 0; r < 8; r++) {
      float tmax = fmaxf(sc[0][r], sc[1][r]);
      for (int off = 8; off >= 1; off >>= 1)
        tmax = fmaxf(tmax, __shfl_xor(tmax, off, 32));
      float nm = fmaxf(rmax[r], tmax);
      float corr = __expf(rmax[r] - nm);
      float p0 = __expf(sc[0][r] - nm);
      float p1 = __expf(sc[1][r] - nm);
      float ps = p0 + p1;
      for (int off = 8; off >= 1; off >>= 1) ps += __shfl_xor(ps, off, 32);
      rsum[r] = rsum[r] * corr + ps;
      rmax[r] = nm;
      for (int t = 0; t < 4; t++) o[t][r] = o[t][r] * corr;
      Ps[wv][r + 8 * grp][lr] = (bf16_t)p0;
      Ps[wv][r + 8 * grp][16 + lr] = (bf16_t)p1;
    }
    v16bf pa = load_frag(&Ps[wv][lr][grp * 8], &Ps[wv][lr][16 + grp * 8]);
    for (int nt = 0; nt < 4; nt++) {
      const bf16_t* vrow = Vt + ((size_t)bh * DHEAD + nt * 16 + lr) * N_IN + kt;
      v16bf bv = load_frag(vrow + grp * 8, vrow + 16 + grp * 8);
      o[nt] = __builtin_amdgcn_wmma_f32_16x16x32_bf16(false, pa, false, bv,
                                                      (short)0, o[nt], false, false);
    }
  }

  const float* lts = lat_ts + (size_t)bb * N_LAT;
  for (int r = 0; r < 8; r++) {
    float inv = 1.0f / rsum[r];
    int m = q0 + r + 8 * grp;
    float t = lts[m];
    for (int nt = 0; nt < 4; nt++) {
      float v = o[nt][r] * inv;
      float pv = __shfl_xor(v, 1, 32);
      int d = nt * 16 + lr, j = d >> 1;
      float c, s;
      __sincosf(t * inv_freq_of(j), &s, &c);
      float out = (d & 1) ? (v * c - pv * s) : (v * c + pv * s);
      Out[((size_t)(bb * N_LAT + m)) * INNER + h * DHEAD + d] = (bf16_t)out;
    }
  }
}

// ---------------------------------------------------------------------------
extern "C" void kernel_launch(void* const* d_in, const int* in_sizes, int n_in,
                              void* d_out, int out_size, void* d_ws, size_t ws_size,
                              hipStream_t stream) {
  (void)in_sizes; (void)n_in; (void)out_size; (void)ws_size;
  const float* inputs  = (const float*)d_in[0];
  const float* latents = (const float*)d_in[1];
  const float* in_ts   = (const float*)d_in[3];
  const float* lat_ts  = (const float*)d_in[4];
  const unsigned char* mask = (const unsigned char*)d_in[6];
  const float* ln_q_w = (const float*)d_in[7];
  const float* ln_q_b = (const float*)d_in[8];
  const float* ln_c_w = (const float*)d_in[9];
  const float* ln_c_b = (const float*)d_in[10];
  const float* w_q  = (const float*)d_in[11];
  const float* w_kv = (const float*)d_in[12];
  const float* w_o  = (const float*)d_in[13];
  const float* b_o  = (const float*)d_in[14];
  const float* ln_f_w = (const float*)d_in[15];
  const float* ln_f_b = (const float*)d_in[16];
  const float* w1 = (const float*)d_in[17];
  const float* b1 = (const float*)d_in[18];
  const float* w2 = (const float*)d_in[19];
  const float* b2 = (const float*)d_in[20];
  float* out = (float*)d_out;

  char* ws = (char*)d_ws;
  size_t off = 0;
  auto alloc = [&](size_t bytes) -> char* {
    char* p = ws + off;
    off += (bytes + 255) & ~(size_t)255;
    return p;
  };
  bf16_t* xq   = (bf16_t*)alloc((size_t)B_ * N_LAT * DIM * 2);
  bf16_t* xc   = (bf16_t*)alloc((size_t)B_ * N_IN * DIM * 2);
  bf16_t* wqT  = (bf16_t*)alloc((size_t)INNER * DIM * 2);
  bf16_t* wkvT = (bf16_t*)alloc((size_t)2 * INNER * DIM * 2);
  bf16_t* woT  = (bf16_t*)alloc((size_t)DIM * INNER * 2);
  bf16_t* w1T  = (bf16_t*)alloc((size_t)2 * FF * DIM * 2);
  bf16_t* w2T  = (bf16_t*)alloc((size_t)DIM * FF * 2);
  bf16_t* qb   = (bf16_t*)alloc((size_t)B_ * HEADS * N_LAT * DHEAD * 2);
  bf16_t* kb   = (bf16_t*)alloc((size_t)B_ * HEADS * N_IN * DHEAD * 2);
  bf16_t* vT   = (bf16_t*)alloc((size_t)B_ * HEADS * DHEAD * N_IN * 2);
  bf16_t* ao   = (bf16_t*)alloc((size_t)B_ * N_LAT * INNER * 2);
  bf16_t* lnf  = (bf16_t*)alloc((size_t)B_ * N_LAT * DIM * 2);
  bf16_t* gat  = (bf16_t*)alloc((size_t)B_ * N_LAT * FF * 2);

  transpose_to_bf16<<<(512 * 512 + 255) / 256, 256, 0, stream>>>(w_q, wqT, 512, 512);
  transpose_to_bf16<<<(512 * 1024 + 255) / 256, 256, 0, stream>>>(w_kv, wkvT, 512, 1024);
  transpose_to_bf16<<<(512 * 512 + 255) / 256, 256, 0, stream>>>(w_o, woT, 512, 512);
  transpose_to_bf16<<<(512 * 4096 + 255) / 256, 256, 0, stream>>>(w1, w1T, 512, 4096);
  transpose_to_bf16<<<(2048 * 512 + 255) / 256, 256, 0, stream>>>(w2, w2T, 2048, 512);

  layernorm_to_bf16<<<B_ * N_LAT, 256, 0, stream>>>(latents, ln_q_w, ln_q_b, xq);
  layernorm_to_bf16<<<B_ * N_IN, 256, 0, stream>>>(inputs, ln_c_w, ln_c_b, xc);

  {  // q = ln(latents) @ w_q, scaled + rotary, split heads
    dim3 g(512 / BN, (B_ * N_LAT) / BM);
    gemm_bf16<<<g, 256, 0, stream>>>(xq, wqT, B_ * N_LAT, 512, 512, 1,
                                     nullptr, nullptr, nullptr, lat_ts, qb, nullptr);
  }
  {  // kv = ln(inputs) @ w_kv, rotary, K row-major + V transposed
    dim3 g(1024 / BN, (B_ * N_IN) / BM);
    gemm_bf16<<<g, 256, 0, stream>>>(xc, wkvT, B_ * N_IN, 1024, 512, 2,
                                     nullptr, nullptr, nullptr, in_ts, kb, vT);
  }
  {  // flash attention + inverse rotary + head merge
    dim3 g(N_LAT / 128, B_ * HEADS);
    attention_kernel<<<g, 256, 0, stream>>>(qb, kb, vT, mask, lat_ts, ao);
  }
  {  // out-proj + b_o + residual(latents) -> lat (in d_out)
    dim3 g(512 / BN, (B_ * N_LAT) / BM);
    gemm_bf16<<<g, 256, 0, stream>>>(ao, woT, B_ * N_LAT, 512, 512, 0,
                                     b_o, latents, out, nullptr, nullptr, nullptr);
  }
  layernorm_to_bf16<<<B_ * N_LAT, 256, 0, stream>>>(out, ln_f_w, ln_f_b, lnf);
  {  // ffn1 gated gelu -> bf16
    dim3 g(FF / BN, (B_ * N_LAT) / BM);
    gemm_ffn1_gated<<<g, 256, 0, stream>>>(lnf, w1T, b1, gat, B_ * N_LAT, 512);
  }
  {  // ffn2 + b2 + residual accumulate into d_out
    dim3 g(512 / BN, (B_ * N_LAT) / BM);
    gemm_bf16<<<g, 256, 0, stream>>>(gat, w2T, B_ * N_LAT, 512, 2048, 0,
                                     b2, out, out, nullptr, nullptr, nullptr);
  }
}
